// ActorGCN_54709293417098
// MI455X (gfx1250) — matrix operand order repaired and checked by reading, not compiled, not run on hardware
//
#include <hip/hip_runtime.h>
#include <hip/hip_bf16.h>
#include <math.h>

typedef __attribute__((ext_vector_type(16))) _Float16 v16h;
typedef __attribute__((ext_vector_type(8)))  _Float16 v8h;
typedef __attribute__((ext_vector_type(8)))  float    v8f;

#define HID 64

// ---------------- init / prep kernels ----------------

__global__ void k_fill_f32(float* p, float v, int n) {
    int i = blockIdx.x * blockDim.x + threadIdx.x;
    if (i < n) p[i] = v;
}

// h0 (f16, stride 32): cols 0..13 = x, col 14 = stem bit (set later), 15..31 = 0
__global__ void k_build_h0(const float* __restrict__ x, _Float16* __restrict__ h0,
                           int N) {
    int gid = blockIdx.x * blockDim.x + threadIdx.x;
    if (gid >= N * 32) return;
    int n = gid >> 5, c = gid & 31;
    float v = (c < 14) ? x[n * 14 + c] : 0.0f;
    h0[gid] = (_Float16)v;
}

__global__ void k_stem_set(const int* __restrict__ stem, _Float16* __restrict__ h0, int S) {
    int i = blockIdx.x * blockDim.x + threadIdx.x;
    if (i < S) h0[stem[i] * 32 + 14] = (_Float16)1.0f;
}

// Pack W[rowsIn x colsIn] (f32) into WMMA B-fragment order (f16, zero padded):
// Bp[((kcIdx*colTiles + tile)*32 + lane)*16 + i]  where
//   k = kcIdx*32 + (i<8 ? 8*(lane>>4) + i : 16 + 8*(lane>>4) + (i-8)) ,  n = tile*16 + (lane&15)
__global__ void k_pack_b(const float* __restrict__ W, _Float16* __restrict__ Bp,
                         int rowsIn, int colsIn, int kChunks, int colTiles) {
    int gid = blockIdx.x * blockDim.x + threadIdx.x;
    int total = kChunks * colTiles * 512;
    if (gid >= total) return;
    int i     = gid & 15;
    int lane  = (gid >> 4) & 31;
    int tile  = (gid >> 9) % colTiles;
    int kcIdx = gid / (512 * colTiles);
    int k = kcIdx * 32 + ((i < 8) ? (8 * (lane >> 4) + i) : (16 + 8 * (lane >> 4) + (i - 8)));
    int n = tile * 16 + (lane & 15);
    float v = (k < rowsIn && n < colsIn) ? W[k * colsIn + n] : 0.0f;
    Bp[gid] = (_Float16)v;
}

__global__ void k_deg_edges(const int* __restrict__ dst, float* __restrict__ deg, int E) {
    int e = blockIdx.x * blockDim.x + threadIdx.x;
    if (e < E) atomicAdd(&deg[dst[e]], 1.0f);
}

__global__ void k_rsqrt(float* __restrict__ d, int N) {
    int i = blockIdx.x * blockDim.x + threadIdx.x;
    if (i < N) d[i] = rsqrtf(d[i]);   // deg >= 1 always (self loop)
}

// ---------------- WMMA GEMM: C = A @ B (+bias) ----------------
// A: f16 [nRows x LDA] row-major, Bp: pre-swizzled fragments (see k_pack_b).
// One wave -> one 16x16 output tile. blockIdx.x = row tile, wave = col tile.
// Rows assumed multiple of 16. KDIM multiple of 32. LDC also = #stored cols.
template <int KDIM, int LDA, int LDC>
__global__ void k_gemm_wmma(const _Float16* __restrict__ A,
                            const _Float16* __restrict__ Bp, int colTiles,
                            const float* __restrict__ bias,
                            float* __restrict__ C) {
    const int lane = threadIdx.x & 31;
    const int wave = threadIdx.x >> 5;
    const int m0 = blockIdx.x * 16;

    const int kAoff = (lane < 16) ? 0 : 8;   // ISA 16-bit A 16x32 layout
    const _Float16* arow = A + (m0 + (lane & 15)) * LDA + kAoff;
    const _Float16* brow = Bp + (wave * 32 + lane) * 16;

    v8f acc = {};
#pragma unroll
    for (int kc = 0; kc < KDIM; kc += 32) {
        v8h alo = *(const v8h*)(arow + kc);
        v8h ahi = *(const v8h*)(arow + kc + 16);
        v16h a = __builtin_shufflevector(alo, ahi, 0, 1, 2, 3, 4, 5, 6, 7,
                                         8, 9, 10, 11, 12, 13, 14, 15);
        v16h b = *(const v16h*)(brow + (kc >> 5) * colTiles * 512);
        acc = __builtin_amdgcn_wmma_f32_16x16x32_f16(false, a, false, b,
                                                     (short)0, acc, false, false);
    }

    const int nB = wave * 16 + (lane & 15);
    if (nB < LDC) {                    // only trims the last head tile (112 -> 105)
        float bv = bias ? bias[nB] : 0.0f;
        float* cp = C + (m0 + ((lane < 16) ? 0 : 8)) * LDC + nB;
#pragma unroll
        for (int v = 0; v < 8; ++v)
            cp[v * LDC] = acc[v] + bv;   // constant-folded immediate offsets
    }
}

// ---------------- edge scatter: agg[dst] += xw[src] * dis[src]*dis[dst] ----------------
// 16 threads per edge, 4 cols each (H=64).
__global__ void k_scatter(const float* __restrict__ xw,
                          const int* __restrict__ src, const int* __restrict__ dst,
                          const float* __restrict__ dis,
                          float* __restrict__ agg, int E) {
    int gid = blockIdx.x * blockDim.x + threadIdx.x;
    int e = gid >> 4;
    if (e >= E) return;
    int c0 = (gid & 15) << 2;
    int s = src[e], d = dst[e];
    float nm = dis[s] * dis[d];
    float4 v = *(const float4*)(xw + s * HID + c0);
    float* ap = agg + d * HID + c0;
    atomicAdd(ap + 0, v.x * nm);
    atomicAdd(ap + 1, v.y * nm);
    atomicAdd(ap + 2, v.z * nm);
    atomicAdd(ap + 3, v.w * nm);
}

// ---------------- self-loop + bias + relu + dropout + f32->f16 ----------------
__global__ void k_post(const float* __restrict__ agg, const float* __restrict__ xw,
                       const float* __restrict__ dis, const float* __restrict__ bias,
                       const float* __restrict__ drop, _Float16* __restrict__ hout, int N) {
    int gid = blockIdx.x * blockDim.x + threadIdx.x;
    if (gid >= N * HID) return;
    int n = gid >> 6, c = gid & 63;
    float di = dis[n];
    float v = agg[gid] + xw[gid] * di * di + bias[c];
    v = fmaxf(v, 0.0f) * drop[gid];
    hout[gid] = (_Float16)v;
}

// ---------------- softmax(logits + g) over 105 cols, one wave per node ----------------
__global__ void k_softmax105(const float* __restrict__ logits, const float* __restrict__ g,
                             float* __restrict__ out, int N) {
    int lane = threadIdx.x & 31;
    int wave = threadIdx.x >> 5;
    int n = blockIdx.x * (blockDim.x >> 5) + wave;
    if (n >= N) return;
    const int base = n * 105;
    float z[4];
    float m = -INFINITY;
#pragma unroll
    for (int j = 0; j < 4; ++j) {
        int c = lane + 32 * j;
        z[j] = (c < 105) ? logits[base + c] + g[base + c] : -INFINITY;
        m = fmaxf(m, z[j]);
    }
    for (int off = 16; off; off >>= 1) m = fmaxf(m, __shfl_xor(m, off, 32));
    float s = 0.0f;
#pragma unroll
    for (int j = 0; j < 4; ++j) {
        int c = lane + 32 * j;
        z[j] = __expf(z[j] - m);
        if (c < 105) s += z[j];
    }
    for (int off = 16; off; off >>= 1) s += __shfl_xor(s, off, 32);
    float inv = 1.0f / s;
#pragma unroll
    for (int j = 0; j < 4; ++j) {
        int c = lane + 32 * j;
        if (c < 105) out[base + c] = z[j] * inv;
    }
}

// ---------------- host side ----------------

extern "C" void kernel_launch(void* const* d_in, const int* in_sizes, int n_in,
                              void* d_out, int out_size, void* d_ws, size_t ws_size,
                              hipStream_t stream) {
    const float* x     = (const float*)d_in[0];
    const float* W1    = (const float*)d_in[1];
    const float* b1    = (const float*)d_in[2];
    const float* W2    = (const float*)d_in[3];
    const float* b2    = (const float*)d_in[4];
    const float* Wb    = (const float*)d_in[5];
    const float* bb    = (const float*)d_in[6];
    const float* Ws    = (const float*)d_in[7];
    const float* bs    = (const float*)d_in[8];
    const float* drop1 = (const float*)d_in[9];
    const float* drop2 = (const float*)d_in[10];
    const float* gb    = (const float*)d_in[11];
    const float* gs    = (const float*)d_in[12];
    const int*   eidx  = (const int*)d_in[13];
    const int*   stem  = (const int*)d_in[14];

    const int N = in_sizes[0] / 14;
    const int E = in_sizes[13] / 2;
    const int S = in_sizes[14];
    const int* src = eidx;
    const int* dst = eidx + E;

    // workspace carve-out (256B aligned)
    char* w = (char*)d_ws;
    size_t off = 0;
    auto carve = [&](size_t bytes) -> void* {
        void* p = w + off;
        off = (off + bytes + 255) & ~(size_t)255;
        return p;
    };
    float*    dis  = (float*)   carve((size_t)N * 4);            // deg -> rsqrt(deg)
    _Float16* h0h  = (_Float16*)carve((size_t)N * 32 * 2);       // layer1 input, K padded to 32
    _Float16* hh   = (_Float16*)carve((size_t)N * HID * 2);      // h1 then h2 (f16)
    float*    xw   = (float*)   carve((size_t)N * HID * 4);
    float*    agg  = (float*)   carve((size_t)N * HID * 4);
    _Float16* W1p  = (_Float16*)carve((size_t)1 * 4 * 512 * 2);  // K=32,  4 col tiles
    _Float16* W2p  = (_Float16*)carve((size_t)2 * 4 * 512 * 2);  // K=64,  4 col tiles
    _Float16* Wbp  = (_Float16*)carve((size_t)2 * 7 * 512 * 2);  // K=64,  7 col tiles
    _Float16* Wsp  = (_Float16*)carve((size_t)2 * 7 * 512 * 2);

    float* out_blk  = (float*)d_out;
    float* out_stm  = out_blk + (size_t)N * 105;
    float* out_selb = out_stm + (size_t)N * 105;
    float* out_sels = out_selb + (size_t)N * 105;

    const int B256 = 256;
    const int rowTiles = (N + 15) / 16;

    // --- degree / norm ---
    k_fill_f32<<<(N + B256 - 1) / B256, B256, 0, stream>>>(dis, 1.0f, N); // self loop
    k_deg_edges<<<(E + B256 - 1) / B256, B256, 0, stream>>>(dst, dis, E);
    k_rsqrt<<<(N + B256 - 1) / B256, B256, 0, stream>>>(dis, N);

    // --- inputs / weights to f16 (B pre-swizzled into WMMA fragment order) ---
    k_build_h0<<<((size_t)N * 32 + B256 - 1) / B256, B256, 0, stream>>>(x, h0h, N);
    k_stem_set<<<(S + B256 - 1) / B256, B256, 0, stream>>>(stem, h0h, S);
    k_pack_b<<<(1 * 4 * 512 + B256 - 1) / B256, B256, 0, stream>>>(W1, W1p, 15, 64, 1, 4);
    k_pack_b<<<(2 * 4 * 512 + B256 - 1) / B256, B256, 0, stream>>>(W2, W2p, 64, 64, 2, 4);
    k_pack_b<<<(2 * 7 * 512 + B256 - 1) / B256, B256, 0, stream>>>(Wb, Wbp, 64, 105, 2, 7);
    k_pack_b<<<(2 * 7 * 512 + B256 - 1) / B256, B256, 0, stream>>>(Ws, Wsp, 64, 105, 2, 7);

    // --- layer 1 ---
    k_gemm_wmma<32, 32, 64><<<rowTiles, 128, 0, stream>>>(h0h, W1p, 4, nullptr, xw);
    k_fill_f32<<<((size_t)N * HID + B256 - 1) / B256, B256, 0, stream>>>(agg, 0.0f, N * HID);
    k_scatter<<<((size_t)E * 16 + B256 - 1) / B256, B256, 0, stream>>>(xw, src, dst, dis, agg, E);
    k_post<<<((size_t)N * HID + B256 - 1) / B256, B256, 0, stream>>>(agg, xw, dis, b1, drop1, hh, N);

    // --- layer 2 ---
    k_gemm_wmma<64, 64, 64><<<rowTiles, 128, 0, stream>>>(hh, W2p, 4, nullptr, xw);
    k_fill_f32<<<((size_t)N * HID + B256 - 1) / B256, B256, 0, stream>>>(agg, 0.0f, N * HID);
    k_scatter<<<((size_t)E * 16 + B256 - 1) / B256, B256, 0, stream>>>(xw, src, dst, dis, agg, E);
    k_post<<<((size_t)N * HID + B256 - 1) / B256, B256, 0, stream>>>(agg, xw, dis, b2, drop2, hh, N);

    // --- heads (logits straight into d_out) ---
    k_gemm_wmma<64, 64, 105><<<rowTiles, 224, 0, stream>>>(hh, Wbp, 7, bb, out_blk);
    k_gemm_wmma<64, 64, 105><<<rowTiles, 224, 0, stream>>>(hh, Wsp, 7, bs, out_stm);

    // --- gumbel softmax (soft) ---
    k_softmax105<<<(N + 7) / 8, 256, 0, stream>>>(out_blk, gb, out_selb, N);
    k_softmax105<<<(N + 7) / 8, 256, 0, stream>>>(out_stm, gs, out_sels, N);
    (void)n_in; (void)out_size; (void)ws_size;
}
